// GroupedMLP_3332894622532
// MI455X (gfx1250) — compile-verified, matrix operand order checked by reading, not run on hardware
//
#include <hip/hip_runtime.h>
#include <cstdint>
#include <cstddef>

#define T_TOK 4096
#define HDIM  2048
#define NEXP  8
#define IDIM  4096

typedef __bf16 v16bf __attribute__((ext_vector_type(16)));
typedef float  v8f   __attribute__((ext_vector_type(8)));

__device__ __forceinline__ unsigned short f2bf(float f) {
  unsigned int x = __float_as_uint(f);
  x += 0x7FFFu + ((x >> 16) & 1u);   // round-to-nearest-even
  return (unsigned short)(x >> 16);
}

// Async global->LDS copy of 16B per lane (tracked with ASYNCcnt).
__device__ __forceinline__ void async_b128(unsigned int lds_off, uint64_t gaddr) {
  asm volatile("global_load_async_to_lds_b128 %0, %1, off"
               :: "v"(lds_off), "v"(gaddr) : "memory");
}
__device__ __forceinline__ void wait_async0() {
  asm volatile("s_wait_asynccnt 0" ::: "memory");
}

__device__ __forceinline__ v8f wmma_bf16(v16bf a, v16bf b, v8f c) {
  return __builtin_amdgcn_wmma_f32_16x16x32_bf16(false, a, false, b, (short)0, c,
                                                 false, false);
}

// ---------------- routing ----------------

__global__ void zero_counts_kernel(int* counts) {
  if (threadIdx.x < NEXP) counts[threadIdx.x] = 0;
}

__global__ void router_kernel(const float* __restrict__ logits,
                              int* __restrict__ ids, float* __restrict__ wts,
                              int* __restrict__ counts) {
  int t = blockIdx.x * blockDim.x + threadIdx.x;
  if (t >= T_TOK) return;
  float l[NEXP];
  float mx = -1e30f;
  for (int e = 0; e < NEXP; ++e) { l[e] = logits[t * NEXP + e]; mx = fmaxf(mx, l[e]); }
  float s = 0.f;
  for (int e = 0; e < NEXP; ++e) { l[e] = __expf(l[e] - mx); s += l[e]; }
  float inv = 1.f / s;
  int i1 = 0; float p1 = -1.f;
  for (int e = 0; e < NEXP; ++e) { float p = l[e] * inv; if (p > p1) { p1 = p; i1 = e; } }
  int i2 = 0; float p2 = -1.f;
  for (int e = 0; e < NEXP; ++e) {
    if (e == i1) continue;
    float p = l[e] * inv; if (p > p2) { p2 = p; i2 = e; }
  }
  ids[t * 2 + 0] = i1; wts[t * 2 + 0] = p1;
  ids[t * 2 + 1] = i2; wts[t * 2 + 1] = p2;
  atomicAdd(&counts[i1], 1);
  atomicAdd(&counts[i2], 1);
}

__global__ void scan_kernel(const int* __restrict__ counts, int* __restrict__ offsets,
                            int* __restrict__ cursor) {
  if (threadIdx.x == 0) {
    int acc = 0;
    for (int e = 0; e < NEXP; ++e) { offsets[e] = acc; cursor[e] = acc; acc += counts[e]; }
    offsets[NEXP] = acc;
  }
}

__global__ void scatter_kernel(const int* __restrict__ ids, const float* __restrict__ wts,
                               int* __restrict__ cursor, int* __restrict__ stok,
                               int* __restrict__ sslot, float* __restrict__ sw) {
  int t = blockIdx.x * blockDim.x + threadIdx.x;
  if (t >= T_TOK) return;
  for (int k = 0; k < 2; ++k) {
    int e = ids[t * 2 + k];
    int pos = atomicAdd(&cursor[e], 1);
    stok[pos] = t;
    sslot[pos] = k;
    sw[pos] = wts[t * 2 + k];
  }
}

__global__ void cvt_x_kernel(const float4* __restrict__ x, ushort4* __restrict__ xb) {
  int i = blockIdx.x * blockDim.x + threadIdx.x;
  if (i >= T_TOK * HDIM / 4) return;
  float4 v = x[i];
  xb[i] = make_ushort4(f2bf(v.x), f2bf(v.y), f2bf(v.z), f2bf(v.w));
}

// ---------------- GEMM1: act = silu(x@Wg^T) * (x@Wu^T), bf16 ----------------
// Block: 128 gathered tokens x 64 intermediate cols (gate+up pair), K=H in
// 32-chunks, double-buffered LDS + software pipelined async A copies.

__global__ __launch_bounds__(256) void gemm1_kernel(
    const unsigned short* __restrict__ xb, const float* __restrict__ w1,
    const int* __restrict__ counts, const int* __restrict__ offsets,
    const int* __restrict__ stok, unsigned short* __restrict__ act) {
  const int e = blockIdx.z;
  const int cnt = counts[e];
  const int mt = blockIdx.x;
  if (mt * 128 >= cnt) return;
  const int base = offsets[e];
  const int ncol0 = blockIdx.y * 64;

  __shared__ __align__(32) unsigned short lds_a[2 * 128 * 32];
  __shared__ __align__(32) unsigned short lds_bg[2 * 64 * 32];
  __shared__ __align__(32) unsigned short lds_bu[2 * 64 * 32];

  const int tid = threadIdx.x;
  const int lane = tid & 31;
  const int wv = tid >> 5;
  const int wm = wv & 3, wn = wv >> 2;        // 4x2 wave grid -> 32x32 per wave
  const int lrow = lane & 15, lhi = lane >> 4;

  // A-tile async copy setup: 512 x b128 chunks, 2 per thread.
  const int ar0 = tid >> 2, ar1 = 64 + (tid >> 2);
  const int apart = tid & 3;
  const int ia0 = (mt * 128 + ar0 < cnt) ? (base + mt * 128 + ar0) : base;
  const int ia1 = (mt * 128 + ar1 < cnt) ? (base + mt * 128 + ar1) : base;
  const long long tok0 = stok[ia0];
  const long long tok1 = stok[ia1];
  const uint64_t xbase = (uint64_t)xb;
  const uint64_t ga0 = xbase + (uint64_t)(tok0 * HDIM + apart * 8) * 2;
  const uint64_t ga1 = xbase + (uint64_t)(tok1 * HDIM + apart * 8) * 2;
  const unsigned int la0 = (unsigned int)(uintptr_t)&lds_a[ar0 * 32 + apart * 8];
  const unsigned int la1 = (unsigned int)(uintptr_t)&lds_a[ar1 * 32 + apart * 8];

  // B-tile load setup: 64 rows x 8 float4 parts per matrix half.
  const int br0 = tid >> 3, br1 = 32 + (tid >> 3);
  const int bpart = tid & 7;
  const float* w1e = w1 + (size_t)e * 2 * IDIM * HDIM;
  const float* g0 = w1e + (size_t)(ncol0 + br0) * HDIM + bpart * 4;
  const float* g1 = w1e + (size_t)(ncol0 + br1) * HDIM + bpart * 4;
  const float* u0 = g0 + (size_t)IDIM * HDIM;
  const float* u1 = g1 + (size_t)IDIM * HDIM;
  unsigned short* sg0 = &lds_bg[br0 * 32 + bpart * 4];
  unsigned short* sg1 = &lds_bg[br1 * 32 + bpart * 4];
  unsigned short* su0 = &lds_bu[br0 * 32 + bpart * 4];
  unsigned short* su1 = &lds_bu[br1 * 32 + bpart * 4];

  v8f accg[2][2], accu[2][2];
  for (int mi = 0; mi < 2; ++mi)
    for (int ni = 0; ni < 2; ++ni) { accg[mi][ni] = (v8f)0.0f; accu[mi][ni] = (v8f)0.0f; }

  // Pipeline prologue: tile 0 in flight.
  float4 fg0 = *(const float4*)(g0);
  float4 fg1 = *(const float4*)(g1);
  float4 fu0 = *(const float4*)(u0);
  float4 fu1 = *(const float4*)(u1);
  async_b128(la0, ga0);
  async_b128(la1, ga1);

  int buf = 0;
  for (int kk = 0; kk < HDIM; kk += 32) {
    const int ao = buf * (128 * 32);
    const int bo = buf * (64 * 32);
    *(ushort4*)(sg0 + bo) = make_ushort4(f2bf(fg0.x), f2bf(fg0.y), f2bf(fg0.z), f2bf(fg0.w));
    *(ushort4*)(sg1 + bo) = make_ushort4(f2bf(fg1.x), f2bf(fg1.y), f2bf(fg1.z), f2bf(fg1.w));
    *(ushort4*)(su0 + bo) = make_ushort4(f2bf(fu0.x), f2bf(fu0.y), f2bf(fu0.z), f2bf(fu0.w));
    *(ushort4*)(su1 + bo) = make_ushort4(f2bf(fu1.x), f2bf(fu1.y), f2bf(fu1.z), f2bf(fu1.w));

    wait_async0();        // current A tile landed in LDS
    __syncthreads();      // stores visible; all reads of the other buffer done

    const int nbuf = buf ^ 1;
    if (kk + 32 < HDIM) {
      // Launch next tile into the other buffer while we compute this one.
      async_b128(la0 + (unsigned)nbuf * (128 * 32 * 2), ga0 + (uint64_t)(kk + 32) * 2);
      async_b128(la1 + (unsigned)nbuf * (128 * 32 * 2), ga1 + (uint64_t)(kk + 32) * 2);
      fg0 = *(const float4*)(g0 + kk + 32);
      fg1 = *(const float4*)(g1 + kk + 32);
      fu0 = *(const float4*)(u0 + kk + 32);
      fu1 = *(const float4*)(u1 + kk + 32);
      if (kk + 64 < HDIM) {
        __builtin_prefetch(g0 + kk + 64, 0, 3);
        __builtin_prefetch(u0 + kk + 64, 0, 3);
      }
    }

    v16bf a0  = *(const v16bf*)&lds_a [ao + (wm * 32 +  0 + lrow) * 32 + lhi * 16];
    v16bf a1  = *(const v16bf*)&lds_a [ao + (wm * 32 + 16 + lrow) * 32 + lhi * 16];
    v16bf bg0 = *(const v16bf*)&lds_bg[bo + (wn * 32 +  0 + lrow) * 32 + lhi * 16];
    v16bf bg1 = *(const v16bf*)&lds_bg[bo + (wn * 32 + 16 + lrow) * 32 + lhi * 16];
    v16bf bu0 = *(const v16bf*)&lds_bu[bo + (wn * 32 +  0 + lrow) * 32 + lhi * 16];
    v16bf bu1 = *(const v16bf*)&lds_bu[bo + (wn * 32 + 16 + lrow) * 32 + lhi * 16];

    accg[0][0] = wmma_bf16(a0, bg0, accg[0][0]);
    accg[0][1] = wmma_bf16(a0, bg1, accg[0][1]);
    accg[1][0] = wmma_bf16(a1, bg0, accg[1][0]);
    accg[1][1] = wmma_bf16(a1, bg1, accg[1][1]);
    accu[0][0] = wmma_bf16(a0, bu0, accu[0][0]);
    accu[0][1] = wmma_bf16(a0, bu1, accu[0][1]);
    accu[1][0] = wmma_bf16(a1, bu0, accu[1][0]);
    accu[1][1] = wmma_bf16(a1, bu1, accu[1][1]);

    buf = nbuf;
  }

  // Epilogue: silu(gate)*up -> bf16 act buffer, row = assignment slot index.
  for (int mi = 0; mi < 2; ++mi) {
    for (int v = 0; v < 8; ++v) {
      int gm = mt * 128 + wm * 32 + mi * 16 + v + 8 * lhi;
      if (gm >= cnt) continue;
      size_t s = (size_t)(base + gm);
      for (int ni = 0; ni < 2; ++ni) {
        int n = ncol0 + wn * 32 + ni * 16 + lrow;
        float g = accg[mi][ni][v];
        float u = accu[mi][ni][v];
        float a = (g / (1.f + __expf(-g))) * u;
        act[s * IDIM + n] = f2bf(a);
      }
    }
  }
}

// ---------------- GEMM2: y = act @ w2^T, scaled & scattered ----------------

__global__ __launch_bounds__(256) void gemm2_kernel(
    const unsigned short* __restrict__ act, const float* __restrict__ w2,
    const int* __restrict__ counts, const int* __restrict__ offsets,
    const int* __restrict__ stok, const int* __restrict__ sslot,
    const float* __restrict__ sw, float* __restrict__ out_slots) {
  const int e = blockIdx.z;
  const int cnt = counts[e];
  const int mt = blockIdx.x;
  if (mt * 128 >= cnt) return;
  const int base = offsets[e];
  const int ncol0 = blockIdx.y * 64;

  __shared__ __align__(32) unsigned short lds_a[2 * 128 * 32];
  __shared__ __align__(32) unsigned short lds_b[2 * 64 * 32];

  const int tid = threadIdx.x;
  const int lane = tid & 31;
  const int wv = tid >> 5;
  const int wm = wv & 3, wn = wv >> 2;
  const int lrow = lane & 15, lhi = lane >> 4;

  const int ar0 = tid >> 2, ar1 = 64 + (tid >> 2);
  const int apart = tid & 3;
  const size_t sa0 = (size_t)((mt * 128 + ar0 < cnt) ? (base + mt * 128 + ar0) : base);
  const size_t sa1 = (size_t)((mt * 128 + ar1 < cnt) ? (base + mt * 128 + ar1) : base);
  const uint64_t abase = (uint64_t)act;
  const uint64_t ga0 = abase + (uint64_t)(sa0 * IDIM + apart * 8) * 2;
  const uint64_t ga1 = abase + (uint64_t)(sa1 * IDIM + apart * 8) * 2;
  const unsigned int la0 = (unsigned int)(uintptr_t)&lds_a[ar0 * 32 + apart * 8];
  const unsigned int la1 = (unsigned int)(uintptr_t)&lds_a[ar1 * 32 + apart * 8];

  const int br0 = tid >> 3, br1 = 32 + (tid >> 3);
  const int bpart = tid & 7;
  const float* w2e = w2 + (size_t)e * HDIM * IDIM;
  const float* b0 = w2e + (size_t)(ncol0 + br0) * IDIM + bpart * 4;
  const float* b1 = w2e + (size_t)(ncol0 + br1) * IDIM + bpart * 4;
  unsigned short* sb0 = &lds_b[br0 * 32 + bpart * 4];
  unsigned short* sb1 = &lds_b[br1 * 32 + bpart * 4];

  v8f acc[2][2];
  for (int mi = 0; mi < 2; ++mi)
    for (int ni = 0; ni < 2; ++ni) acc[mi][ni] = (v8f)0.0f;

  float4 f0 = *(const float4*)(b0);
  float4 f1 = *(const float4*)(b1);
  async_b128(la0, ga0);
  async_b128(la1, ga1);

  int buf = 0;
  for (int kk = 0; kk < IDIM; kk += 32) {
    const int ao = buf * (128 * 32);
    const int bo = buf * (64 * 32);
    *(ushort4*)(sb0 + bo) = make_ushort4(f2bf(f0.x), f2bf(f0.y), f2bf(f0.z), f2bf(f0.w));
    *(ushort4*)(sb1 + bo) = make_ushort4(f2bf(f1.x), f2bf(f1.y), f2bf(f1.z), f2bf(f1.w));

    wait_async0();
    __syncthreads();

    const int nbuf = buf ^ 1;
    if (kk + 32 < IDIM) {
      async_b128(la0 + (unsigned)nbuf * (128 * 32 * 2), ga0 + (uint64_t)(kk + 32) * 2);
      async_b128(la1 + (unsigned)nbuf * (128 * 32 * 2), ga1 + (uint64_t)(kk + 32) * 2);
      f0 = *(const float4*)(b0 + kk + 32);
      f1 = *(const float4*)(b1 + kk + 32);
      if (kk + 64 < IDIM) __builtin_prefetch(b0 + kk + 64, 0, 3);
    }

    v16bf a0  = *(const v16bf*)&lds_a[ao + (wm * 32 +  0 + lrow) * 32 + lhi * 16];
    v16bf a1  = *(const v16bf*)&lds_a[ao + (wm * 32 + 16 + lrow) * 32 + lhi * 16];
    v16bf bb0 = *(const v16bf*)&lds_b[bo + (wn * 32 +  0 + lrow) * 32 + lhi * 16];
    v16bf bb1 = *(const v16bf*)&lds_b[bo + (wn * 32 + 16 + lrow) * 32 + lhi * 16];

    acc[0][0] = wmma_bf16(a0, bb0, acc[0][0]);
    acc[0][1] = wmma_bf16(a0, bb1, acc[0][1]);
    acc[1][0] = wmma_bf16(a1, bb0, acc[1][0]);
    acc[1][1] = wmma_bf16(a1, bb1, acc[1][1]);

    buf = nbuf;
  }

  for (int mi = 0; mi < 2; ++mi) {
    for (int v = 0; v < 8; ++v) {
      int gm = mt * 128 + wm * 32 + mi * 16 + v + 8 * lhi;
      if (gm >= cnt) continue;
      int s = base + gm;
      int tok = stok[s];
      int slot = sslot[s];
      float wgt = sw[s];
      for (int ni = 0; ni < 2; ++ni) {
        int n = ncol0 + wn * 32 + ni * 16 + lrow;
        out_slots[((size_t)tok * 2 + slot) * HDIM + n] = wgt * acc[mi][ni][v];
      }
    }
  }
}

__global__ void combine_kernel(const float* __restrict__ os, float* __restrict__ out) {
  int i = blockIdx.x * blockDim.x + threadIdx.x;
  if (i >= T_TOK * HDIM) return;
  int t = i / HDIM, h = i - t * HDIM;
  size_t b = (size_t)t * 2 * HDIM + h;
  out[i] = os[b] + os[b + HDIM];
}

// ---------------- host launcher ----------------

extern "C" void kernel_launch(void* const* d_in, const int* in_sizes, int n_in,
                              void* d_out, int out_size, void* d_ws, size_t ws_size,
                              hipStream_t stream) {
  const float* hs     = (const float*)d_in[0];
  const float* logits = (const float*)d_in[1];
  const float* w1     = (const float*)d_in[2];
  const float* w2     = (const float*)d_in[3];
  float* out = (float*)d_out;
  (void)hs; (void)in_sizes; (void)n_in; (void)out_size; (void)ws_size;

  char* p = (char*)d_ws;
  auto alloc = [&](size_t bytes) -> void* {
    void* r = (void*)p;
    p += (bytes + 255) & ~(size_t)255;
    return r;
  };
  int*   counts  = (int*)alloc(NEXP * sizeof(int));
  int*   offsets = (int*)alloc((NEXP + 1) * sizeof(int));
  int*   cursor  = (int*)alloc(NEXP * sizeof(int));
  int*   tids    = (int*)alloc((size_t)T_TOK * 2 * sizeof(int));
  float* tw      = (float*)alloc((size_t)T_TOK * 2 * sizeof(float));
  int*   stok    = (int*)alloc((size_t)T_TOK * 2 * sizeof(int));
  int*   sslot   = (int*)alloc((size_t)T_TOK * 2 * sizeof(int));
  float* sw      = (float*)alloc((size_t)T_TOK * 2 * sizeof(float));
  unsigned short* xb  = (unsigned short*)alloc((size_t)T_TOK * HDIM * 2);
  unsigned short* act = (unsigned short*)alloc((size_t)T_TOK * 2 * IDIM * 2);
  float* oslots  = (float*)alloc((size_t)T_TOK * 2 * HDIM * sizeof(float));

  zero_counts_kernel<<<1, 32, 0, stream>>>(counts);
  router_kernel<<<T_TOK / 256, 256, 0, stream>>>(logits, tids, tw, counts);
  scan_kernel<<<1, 32, 0, stream>>>(counts, offsets, cursor);
  scatter_kernel<<<T_TOK / 256, 256, 0, stream>>>(tids, tw, cursor, stok, sslot, sw);
  cvt_x_kernel<<<(T_TOK * HDIM / 4) / 256, 256, 0, stream>>>((const float4*)hs,
                                                             (ushort4*)xb);
  gemm1_kernel<<<dim3(T_TOK * 2 / 128, IDIM / 64, NEXP), 256, 0, stream>>>(
      xb, w1, counts, offsets, stok, act);
  gemm2_kernel<<<dim3(T_TOK * 2 / 128, HDIM / 64, NEXP), 256, 0, stream>>>(
      act, w2, counts, offsets, stok, sslot, sw, oslots);
  combine_kernel<<<(T_TOK * HDIM) / 256, 256, 0, stream>>>(oslots, out);
}
